// Tacotron2_84353157693568
// MI455X (gfx1250) — matrix operand order repaired and checked by reading, not compiled, not run on hardware
//
#include <hip/hip_runtime.h>
#include <hip/hip_bf16.h>
#include <stdint.h>

// ---------------------------------------------------------------------------
// Tacotron2 forward for MI455X (gfx1250).  All dense GEMMs use
// V_WMMA_F32_16X16X32_BF16 (bf16 inputs, fp32 accumulate).  Encoder BiLSTM is
// a persistent one-workgroup-per-direction kernel keeping gates/h/c in the
// 320KB/WGP LDS.  Decoder is a host loop of per-step kernels (latency-bound
// recurrence; decoder weights stay resident in the 192MB L2 as bf16).
// K-loops are software-pipelined: loads for chunk k+32 are issued before the
// WMMA of chunk k, so waves keep 4 global_load_b128 in flight across each
// v_wmma instead of draining LOADcnt to zero.
// ---------------------------------------------------------------------------

#define B_  32
#define S_  256
#define T_  800
#define ENC_ 512
#define HALF_ 256
#define DEC_ 1024
#define ATT_ 128
#define MEL_ 80
#define MELP_ 96          // MEL padded to multiple of 32 for WMMA K

typedef __hip_bfloat16 bhalf;
typedef __attribute__((ext_vector_type(16))) __bf16 v16bf;
typedef __attribute__((ext_vector_type(8)))  float  v8f;

struct U32x8 { uint4 a, b; };
static_assert(sizeof(U32x8) == sizeof(v16bf), "frag size");

// A-matrix fragment (16x32 bf16, row major, lda elements between rows).
// lane<16: row=lane,    K = k0+0..7  (elems 0..7)  and k0+16..23 (elems 8..15)
// lane>=16: row=lane-16, K = k0+8..15 (elems 0..7) and k0+24..31 (elems 8..15)
static __device__ __forceinline__ v16bf load_a_frag(const bhalf* A, int lda, int k0) {
  int lane = threadIdx.x & 31;
  int row  = lane & 15;
  int kb   = (lane < 16) ? 0 : 8;
  const bhalf* p = A + (size_t)row * lda + k0 + kb;
  U32x8 t;
  t.a = *(const uint4*)(p);
  t.b = *(const uint4*)(p + 16);
  return __builtin_bit_cast(v16bf, t);
}

// B-matrix fragment from weight stored [N,K] row major (ldw = K stride).
// lane<16: col=lane, K = k0+0..7, k0+16..23 ; lane>=16: col=lane-16, K = k0+8..15, k0+24..31
static __device__ __forceinline__ v16bf load_b_frag(const bhalf* W, int ldw, int k0) {
  int lane = threadIdx.x & 31;
  int col  = lane & 15;
  int kb   = (lane < 16) ? 0 : 8;
  const bhalf* p = W + (size_t)col * ldw + k0 + kb;
  U32x8 t;
  t.a = *(const uint4*)(p);
  t.b = *(const uint4*)(p + 16);
  return __builtin_bit_cast(v16bf, t);
}

static __device__ __forceinline__ v8f wmma_bf16(v16bf a, v16bf b, v8f acc) {
  return __builtin_amdgcn_wmma_f32_16x16x32_bf16(false, a, false, b, (short)0, acc,
                                                 false, false);
}

// Software-pipelined K-loop: next chunk's loads are in program order BEFORE the
// WMMA consuming the current chunk, so the wave keeps loads in flight across
// each matrix op (s_wait_loadcnt settles at 4 instead of 0).
static __device__ __forceinline__ v8f gemm_loop(const bhalf* __restrict__ Ap, int lda,
                                                const bhalf* __restrict__ Wp, int ldw,
                                                int K, v8f acc) {
  __builtin_prefetch(Wp, 0, 3);
  v16bf a = load_a_frag(Ap, lda, 0);
  v16bf b = load_b_frag(Wp, ldw, 0);
  for (int k0 = 32; k0 < K; k0 += 32) {
    v16bf an = load_a_frag(Ap, lda, k0);
    v16bf bn = load_b_frag(Wp, ldw, k0);
    acc = wmma_bf16(a, b, acc);
    a = an;
    b = bn;
  }
  return wmma_bf16(a, b, acc);
}

// C/D layout: element e -> row ((lane<16)?e:8+e), col = lane&15
static __device__ __forceinline__ void store_tile(float* C, int ldc,
                                                  bhalf* Cb, int ldcb,
                                                  const float* bias16, v8f acc, int act) {
  int lane = threadIdx.x & 31;
  int col  = lane & 15;
  int r0   = (lane < 16) ? 0 : 8;
  float bv = bias16 ? bias16[col] : 0.0f;
#pragma unroll
  for (int e = 0; e < 8; ++e) {
    float v = acc[e] + bv;
    if (act == 1) v = fmaxf(v, 0.0f);
    int r = r0 + e;
    if (C)  C[(size_t)r * ldc + col]  = v;
    if (Cb) Cb[(size_t)r * ldcb + col] = __float2bfloat16(v);
  }
}

static __device__ __forceinline__ float sigf(float x) { return 1.0f / (1.0f + expf(-x)); }

// ---------------------------------------------------------------------------
// Generic dual-input WMMA GEMM: C = act(A1 @ W1^T + [A2 @ W2^T] + bias)
// One 16x16 tile per wave; 4 waves per 128-thread block.
// ---------------------------------------------------------------------------
__global__ void gemm2_kernel(const bhalf* __restrict__ A1, int lda1,
                             const bhalf* __restrict__ W1, int ldw1, int K1,
                             const bhalf* __restrict__ A2, int lda2,
                             const bhalf* __restrict__ W2, int ldw2, int K2,
                             const float* __restrict__ bias,
                             float* C, int ldc, bhalf* Cb, int ldcb,
                             int Mtiles, int Ntiles, int act) {
  int tile = blockIdx.x * (blockDim.x >> 5) + (threadIdx.x >> 5);
  if (tile >= Mtiles * Ntiles) return;
  int tm = tile / Ntiles, tn = tile % Ntiles;
  v8f acc = {0.f,0.f,0.f,0.f,0.f,0.f,0.f,0.f};
  acc = gemm_loop(A1 + (size_t)tm * 16 * lda1, lda1,
                  W1 + (size_t)tn * 16 * ldw1, ldw1, K1, acc);
  if (A2) {
    acc = gemm_loop(A2 + (size_t)tm * 16 * lda2, lda2,
                    W2 + (size_t)tn * 16 * ldw2, ldw2, K2, acc);
  }
  float* Cp  = C  ? (C  + (size_t)tm * 16 * ldc  + tn * 16) : nullptr;
  bhalf* Cbp = Cb ? (Cb + (size_t)tm * 16 * ldcb + tn * 16) : nullptr;
  store_tile(Cp, ldc, Cbp, ldcb, bias ? (bias + tn * 16) : nullptr, acc, act);
}

// ---------------------------------------------------------------------------
// Persistent bidirectional LSTM layer: grid=2 blocks (fwd/bwd), 256 threads.
// Dynamic LDS holds gates[32][1024] f32, c[32][256] f32, h[32][256] bf16.
// ---------------------------------------------------------------------------
__global__ void bilstm_layer_kernel(const bhalf* __restrict__ xs,        // [S,B,512] bf16
                                    const bhalf* WihF, const bhalf* WhhF, const float* bgF,
                                    const bhalf* WihB, const bhalf* WhhB, const float* bgB,
                                    bhalf* outB,                          // [S,B,512] bf16
                                    float* outF,                          // nullable [S,B,512]
                                    int S, int Bn) {
  extern __shared__ char smem[];
  float* gates = (float*)smem;                       // 32*1024 f32 = 128KB
  float* cst   = gates + 32 * 1024;                  // 32*256  f32 = 32KB
  bhalf* hst   = (bhalf*)(cst + 32 * 256);           // 32*256  bf16 = 16KB
  int dir = blockIdx.x;
  const bhalf* Wih = dir ? WihB : WihF;
  const bhalf* Whh = dir ? WhhB : WhhF;
  const float* bg  = dir ? bgB  : bgF;
  int colofs = dir ? HALF_ : 0;
  int tid = threadIdx.x;
  for (int i = tid; i < 32 * 256; i += blockDim.x) { cst[i] = 0.f; hst[i] = __float2bfloat16(0.f); }
  __syncthreads();
  int wave = tid >> 5, nw = blockDim.x >> 5;
  for (int step = 0; step < S; ++step) {
    int t = dir ? (S - 1 - step) : step;
    const bhalf* xt = xs + (size_t)t * Bn * ENC_;
    for (int tile = wave; tile < 2 * 64; tile += nw) {  // gates 32x1024
      int tm = tile >> 6, tn = tile & 63;
      v8f acc = {0.f,0.f,0.f,0.f,0.f,0.f,0.f,0.f};
      acc = gemm_loop(xt + (size_t)tm * 16 * ENC_, ENC_,
                      Wih + (size_t)tn * 16 * ENC_, ENC_, ENC_, acc);
      acc = gemm_loop(hst + (size_t)tm * 16 * HALF_, HALF_,
                      Whh + (size_t)tn * 16 * HALF_, HALF_, HALF_, acc);
      store_tile(gates + (size_t)tm * 16 * 1024 + tn * 16, 1024, nullptr, 0,
                 bg + tn * 16, acc, 0);
    }
    __syncthreads();
    for (int i = tid; i < 32 * 256; i += blockDim.x) {
      int b = i >> 8, j = i & 255;
      const float* g = gates + (size_t)b * 1024;
      float ig = sigf(g[j]), fg = sigf(g[256 + j]);
      float gg = tanhf(g[512 + j]), og = sigf(g[768 + j]);
      float cn = fg * cst[i] + ig * gg;
      float hn = og * tanhf(cn);
      cst[i] = cn;
      hst[i] = __float2bfloat16(hn);
      size_t o = ((size_t)t * Bn + b) * ENC_ + colofs + j;
      outB[o] = __float2bfloat16(hn);
      if (outF) outF[o] = hn;
    }
    __syncthreads();
  }
}

// ---------------------------------------------------------------------------
// Small helper kernels
// ---------------------------------------------------------------------------
__global__ void convert_pad_kernel(bhalf* dst, const float* src, long rows, int scols, int dcols) {
  long i = (long)blockIdx.x * blockDim.x + threadIdx.x;
  if (i >= rows * (long)dcols) return;
  int c = (int)(i % dcols);
  long r = i / dcols;
  dst[i] = __float2bfloat16(c < scols ? src[r * scols + c] : 0.0f);
}

__global__ void xs_transpose_kernel(bhalf* dst, const float* src, int S, int Bn, int C) {
  long i = (long)blockIdx.x * blockDim.x + threadIdx.x;
  if (i >= (long)S * Bn * C) return;
  int c = (int)(i % C);
  long r = i / C;
  int s = (int)(r % S);
  int b = (int)(r / S);
  dst[((size_t)s * Bn + b) * C + c] = __float2bfloat16(src[i]);
}

__global__ void bias_add_kernel(float* dst, const float* a, const float* b, int n) {
  int i = blockIdx.x * blockDim.x + threadIdx.x;
  if (i < n) dst[i] = a[i] + b[i];
}

__global__ void bn_prep_kernel(float* sc, float* sh, const float* bias, const float* gamma,
                               const float* beta, const float* mean, const float* var, int C) {
  int i = blockIdx.x * blockDim.x + threadIdx.x;
  if (i >= C) return;
  float s = gamma[i] * rsqrtf(var[i] + 1e-5f);
  sc[i] = s;
  sh[i] = (bias[i] - mean[i]) * s + beta[i];
}

// conv weight [Cout,Cin,5] f32 -> [5][Cout][Cinp] bf16 (Cin zero-padded)
__global__ void conv_w_convert_kernel(bhalf* dst, const float* src, int Cout, int Cin, int Cinp) {
  long i = (long)blockIdx.x * blockDim.x + threadIdx.x;
  if (i >= 5L * Cout * Cinp) return;
  int ci = (int)(i % Cinp);
  long r = i / Cinp;
  int co = (int)(r % Cout);
  int k  = (int)(r / Cout);
  dst[i] = __float2bfloat16(ci < Cin ? src[((size_t)co * Cin + ci) * 5 + k] : 0.0f);
}

__global__ void lstm_update_kernel(const float* __restrict__ gates, float* __restrict__ c,
                                   bhalf* __restrict__ hB, int ldh, int H, int rows) {
  int i = blockIdx.x * blockDim.x + threadIdx.x;
  if (i >= rows * H) return;
  int b = i / H, j = i % H;
  const float* g = gates + (size_t)b * 4 * H;
  float ig = sigf(g[j]), fg = sigf(g[H + j]);
  float gg = tanhf(g[2 * H + j]), og = sigf(g[3 * H + j]);
  float cn = fg * c[i] + ig * gg;
  c[i] = cn;
  hB[(size_t)b * ldh + j] = __float2bfloat16(og * tanhf(cn));
}

// energies[b,s] = vb + sum_ch v[ch]*tanh(q[b,ch] + proc_mem[s,b,ch] + locconv(b,s,ch))
__global__ void att_energy_kernel(const float* __restrict__ q, const float* __restrict__ pm,
                                  const float* __restrict__ aprev,
                                  const float* __restrict__ locW, const float* __restrict__ locb,
                                  const float* __restrict__ vW, const float* __restrict__ vb,
                                  float* __restrict__ energ, int S, int Bn) {
  int b = blockIdx.x / S, s = blockIdx.x % S;
  int ch = threadIdx.x;  // 128
  float pl = locb[ch];
  const float* wrow = locW + ch * 31;
  for (int j = 0; j < 31; ++j) {
    int sp = s + j - 15;
    if (sp >= 0 && sp < S) pl += wrow[j] * aprev[(size_t)b * S + sp];
  }
  float e = tanhf(q[(size_t)b * ATT_ + ch] + pm[((size_t)s * Bn + b) * ATT_ + ch] + pl) * vW[ch];
  __shared__ float red[128];
  red[ch] = e;
  __syncthreads();
  for (int o = 64; o > 0; o >>= 1) { if (ch < o) red[ch] += red[ch + o]; __syncthreads(); }
  if (ch == 0) energ[(size_t)b * S + s] = red[0] + vb[0];
}

__global__ void softmax_ctx_kernel(const float* __restrict__ energ, const float* __restrict__ memF,
                                   float* __restrict__ aprev, bhalf* attIn, bhalf* decIn,
                                   bhalf* projIn, int S, int Bn) {
  int b = blockIdx.x, tid = threadIdx.x;  // 256 threads, S==256
  __shared__ float aw[256];
  __shared__ float red[256];
  float e = energ[(size_t)b * S + tid];
  red[tid] = e;
  __syncthreads();
  for (int o = 128; o > 0; o >>= 1) { if (tid < o) red[tid] = fmaxf(red[tid], red[tid + o]); __syncthreads(); }
  float mx = red[0];
  __syncthreads();
  float ex = expf(e - mx);
  red[tid] = ex;
  __syncthreads();
  for (int o = 128; o > 0; o >>= 1) { if (tid < o) red[tid] += red[tid + o]; __syncthreads(); }
  float a = ex / red[0];
  aw[tid] = a;
  aprev[(size_t)b * S + tid] = a;
  __syncthreads();
  for (int ch = tid; ch < ENC_; ch += blockDim.x) {
    float acc = 0.f;
    for (int s = 0; s < S; ++s) acc += aw[s] * memF[((size_t)s * Bn + b) * ENC_ + ch];
    bhalf cb = __float2bfloat16(acc);
    attIn[(size_t)b * 640 + 128 + ch]   = cb;
    decIn[(size_t)b * 1536 + 1024 + ch] = cb;
    projIn[(size_t)b * 1536 + 1024 + ch] = cb;
  }
}

__global__ void stop_kernel(const bhalf* __restrict__ projIn, const float* __restrict__ Wst,
                            const float* __restrict__ bst, float* __restrict__ out, int t, int T) {
  int b = threadIdx.x;
  if (b >= B_) return;
  float acc = bst[0];
  for (int i = 0; i < 1536; ++i)
    acc += __bfloat162float(projIn[(size_t)b * 1536 + i]) * Wst[i];
  out[(size_t)b * T + t] = acc;
}

// Postnet conv1d (k=5) as 5 shifted WMMA GEMMs + BN (+tanh) epilogue.
// Xp: time-padded activations [B, T+4, Cinp] bf16; Wk: [5][Cout][Cinp] bf16.
__global__ void postnet_conv_kernel(const bhalf* __restrict__ Xp, int Cinp,
                                    const bhalf* __restrict__ Wk,
                                    const float* __restrict__ sc, const float* __restrict__ sh,
                                    bhalf* Yp, int Coutp,
                                    float* Yf, const float* __restrict__ resid,
                                    int Bn, int T, int Cout, int acttanh) {
  int Ntiles = Cout >> 4;
  int TT = T >> 4;
  int tile = blockIdx.x * (blockDim.x >> 5) + (threadIdx.x >> 5);
  if (tile >= Bn * TT * Ntiles) return;
  int tn = tile % Ntiles;
  int tm = tile / Ntiles;
  int b  = tm / TT;
  int t0 = (tm % TT) << 4;
  v8f acc = {0.f,0.f,0.f,0.f,0.f,0.f,0.f,0.f};
  for (int k = 0; k < 5; ++k) {
    acc = gemm_loop(Xp + ((size_t)(b * (T + 4) + t0 + k)) * Cinp, Cinp,
                    Wk + ((size_t)k * Cout + tn * 16) * Cinp, Cinp, Cinp, acc);
  }
  int lane = threadIdx.x & 31;
  int col  = tn * 16 + (lane & 15);
  int r0   = (lane < 16) ? 0 : 8;
#pragma unroll
  for (int e = 0; e < 8; ++e) {
    int tt = t0 + r0 + e;
    float v = acc[e] * sc[col] + sh[col];
    if (acttanh) v = tanhf(v);
    if (Yp) Yp[((size_t)(b * (T + 4) + tt + 2)) * Coutp + col] = __float2bfloat16(v);
    if (Yf) {
      size_t o = ((size_t)(b * T + tt)) * Cout + col;
      Yf[o] = v + resid[o];
    }
  }
}

// ---------------------------------------------------------------------------
// Host side
// ---------------------------------------------------------------------------
static inline unsigned cdivu(unsigned a, unsigned b) { return (a + b - 1) / b; }

// d_in flattening (setup_inputs dict traversal order):
enum {
  I_EMB = 0, I_MELIN = 1, I_ENCW = 2, I_ENCB = 3,
  I_LSTM0 = 4,   // +d*4 (d: L0f,L0b,L1f,L1b): Wih,Whh,bih,bhh
  I_PRE1W = 20, I_PRE1B = 21, I_PRE2W = 22, I_PRE2B = 23,
  I_ATT_WIH = 24, I_ATT_WHH = 25, I_ATT_BIH = 26, I_ATT_BHH = 27,
  I_QW = 28, I_QB = 29, I_MW = 30, I_MB = 31,
  I_LOCW = 32, I_LOCB = 33, I_VW = 34, I_VB = 35,
  I_DEC_WIH = 36, I_DEC_WHH = 37, I_DEC_BIH = 38, I_DEC_BHH = 39,
  I_MELW = 40, I_MELB = 41, I_STOPW = 42, I_STOPB = 43,
  I_POST0 = 44,  // +i*6: W,b,gamma,beta,mean,var
};

extern "C" void kernel_launch(void* const* d_in, const int* in_sizes, int n_in,
                              void* d_out, int out_size, void* d_ws, size_t ws_size,
                              hipStream_t stream) {
  (void)in_sizes; (void)n_in; (void)out_size; (void)ws_size;
  auto F = [&](int i) { return (const float*)d_in[i]; };
  float* out = (float*)d_out;

  // ---- workspace layout (recomputed identically every call) ----
  char* w = (char*)d_ws;
  auto alloc = [&](size_t bytes) -> void* {
    void* p = (void*)w;
    w += (bytes + 255) & ~(size_t)255;
    return p;
  };
  // bf16 weights
  bhalf* encW    = (bhalf*)alloc((size_t)512 * 512 * 2);
  bhalf* lWih[4]; bhalf* lWhh[4]; float* lBg[4];
  for (int d = 0; d < 4; ++d) {
    lWih[d] = (bhalf*)alloc((size_t)1024 * 512 * 2);
    lWhh[d] = (bhalf*)alloc((size_t)1024 * 256 * 2);
    lBg[d]  = (float*)alloc((size_t)1024 * 4);
  }
  bhalf* pre1W  = (bhalf*)alloc((size_t)256 * MELP_ * 2);
  bhalf* pre2W  = (bhalf*)alloc((size_t)128 * 256 * 2);
  bhalf* attWih = (bhalf*)alloc((size_t)4096 * 640 * 2);
  bhalf* attWhh = (bhalf*)alloc((size_t)4096 * 1024 * 2);
  float* attBg  = (float*)alloc((size_t)4096 * 4);
  bhalf* qW     = (bhalf*)alloc((size_t)128 * 1024 * 2);
  bhalf* mW     = (bhalf*)alloc((size_t)128 * 512 * 2);
  bhalf* decWih = (bhalf*)alloc((size_t)4096 * 1536 * 2);
  bhalf* decWhh = (bhalf*)alloc((size_t)4096 * 1024 * 2);
  float* decBg  = (float*)alloc((size_t)4096 * 4);
  bhalf* melW   = (bhalf*)alloc((size_t)80 * 1536 * 2);
  const int pCin[5]  = {MEL_, 512, 512, 512, 512};
  const int pCinp[5] = {MELP_, 512, 512, 512, 512};
  const int pCout[5] = {512, 512, 512, 512, MEL_};
  bhalf* postW[5]; float* postSc[5]; float* postSh[5];
  for (int i = 0; i < 5; ++i) {
    postW[i]  = (bhalf*)alloc((size_t)5 * pCout[i] * pCinp[i] * 2);
    postSc[i] = (float*)alloc((size_t)pCout[i] * 4);
    postSh[i] = (float*)alloc((size_t)pCout[i] * 4);
  }
  // activations
  const long NBS = (long)B_ * S_;                 // 8192
  bhalf* embB   = (bhalf*)alloc((size_t)NBS * 512 * 2);
  float* xlin   = (float*)alloc((size_t)NBS * 512 * 4);
  bhalf* xs0    = (bhalf*)alloc((size_t)NBS * 512 * 2);
  bhalf* xs1    = (bhalf*)alloc((size_t)NBS * 512 * 2);
  bhalf* memB   = (bhalf*)alloc((size_t)NBS * 512 * 2);
  float* memF   = (float*)alloc((size_t)NBS * 512 * 4);
  float* procM  = (float*)alloc((size_t)NBS * 128 * 4);
  bhalf* melInB = (bhalf*)alloc((size_t)B_ * T_ * MELP_ * 2);
  bhalf* p1B    = (bhalf*)alloc((size_t)32 * 256 * 2);
  bhalf* attIn  = (bhalf*)alloc((size_t)32 * 640 * 2);
  bhalf* decIn  = (bhalf*)alloc((size_t)32 * 1536 * 2);
  bhalf* projIn = (bhalf*)alloc((size_t)32 * 1536 * 2);
  float* gates  = (float*)alloc((size_t)32 * 4096 * 4);
  float* cAtt   = (float*)alloc((size_t)32 * 1024 * 4);
  float* cDec   = (float*)alloc((size_t)32 * 1024 * 4);
  float* qB     = (float*)alloc((size_t)32 * 128 * 4);
  float* energ  = (float*)alloc((size_t)32 * 256 * 4);
  float* aprev  = (float*)alloc((size_t)32 * 256 * 4);
  bhalf* Xp0    = (bhalf*)alloc((size_t)B_ * (T_ + 4) * MELP_ * 2);
  bhalf* Xp1    = (bhalf*)alloc((size_t)B_ * (T_ + 4) * 512 * 2);
  bhalf* Xp2    = (bhalf*)alloc((size_t)B_ * (T_ + 4) * 512 * 2);

  // ---- zero-init carried state / padded buffers ----
  hipMemsetAsync(attIn,  0, (size_t)32 * 640 * 2, stream);
  hipMemsetAsync(decIn,  0, (size_t)32 * 1536 * 2, stream);
  hipMemsetAsync(projIn, 0, (size_t)32 * 1536 * 2, stream);
  hipMemsetAsync(cAtt,   0, (size_t)32 * 1024 * 4, stream);
  hipMemsetAsync(cDec,   0, (size_t)32 * 1024 * 4, stream);
  hipMemsetAsync(aprev,  0, (size_t)32 * 256 * 4, stream);
  hipMemsetAsync(Xp0,    0, (size_t)B_ * (T_ + 4) * MELP_ * 2, stream);
  hipMemsetAsync(Xp1,    0, (size_t)B_ * (T_ + 4) * 512 * 2, stream);
  hipMemsetAsync(Xp2,    0, (size_t)B_ * (T_ + 4) * 512 * 2, stream);

  // ---- weight conversion / folding ----
  auto cvt = [&](bhalf* dst, const float* src, long rows, int sc, int dc) {
    long tot = rows * dc;
    convert_pad_kernel<<<cdivu((unsigned)tot, 256), 256, 0, stream>>>(dst, src, rows, sc, dc);
  };
  cvt(encW, F(I_ENCW), 512, 512, 512);
  for (int d = 0; d < 4; ++d) {
    int base = I_LSTM0 + d * 4;
    cvt(lWih[d], F(base + 0), 1024, 512, 512);
    cvt(lWhh[d], F(base + 1), 1024, 256, 256);
    bias_add_kernel<<<cdivu(1024, 256), 256, 0, stream>>>(lBg[d], F(base + 2), F(base + 3), 1024);
  }
  cvt(pre1W, F(I_PRE1W), 256, MEL_, MELP_);
  cvt(pre2W, F(I_PRE2W), 128, 256, 256);
  cvt(attWih, F(I_ATT_WIH), 4096, 640, 640);
  cvt(attWhh, F(I_ATT_WHH), 4096, 1024, 1024);
  bias_add_kernel<<<cdivu(4096, 256), 256, 0, stream>>>(attBg, F(I_ATT_BIH), F(I_ATT_BHH), 4096);
  cvt(qW, F(I_QW), 128, 1024, 1024);
  cvt(mW, F(I_MW), 128, 512, 512);
  cvt(decWih, F(I_DEC_WIH), 4096, 1536, 1536);
  cvt(decWhh, F(I_DEC_WHH), 4096, 1024, 1024);
  bias_add_kernel<<<cdivu(4096, 256), 256, 0, stream>>>(decBg, F(I_DEC_BIH), F(I_DEC_BHH), 4096);
  cvt(melW, F(I_MELW), 80, 1536, 1536);
  for (int i = 0; i < 5; ++i) {
    int base = I_POST0 + i * 6;
    long tot = 5L * pCout[i] * pCinp[i];
    conv_w_convert_kernel<<<cdivu((unsigned)tot, 256), 256, 0, stream>>>(
        postW[i], F(base + 0), pCout[i], pCin[i], pCinp[i]);
    bn_prep_kernel<<<cdivu((unsigned)pCout[i], 256), 256, 0, stream>>>(
        postSc[i], postSh[i], F(base + 1), F(base + 2), F(base + 3), F(base + 4), F(base + 5),
        pCout[i]);
  }
  cvt(embB, F(I_EMB), NBS, 512, 512);
  cvt(melInB, F(I_MELIN), (long)B_ * T_, MEL_, MELP_);

  // ---- encoder ----
  // x = relu(emb @ encW^T + b) : [8192,512]
  gemm2_kernel<<<cdivu(512u * 32u, 4), 128, 0, stream>>>(
      embB, 512, encW, 512, 512, nullptr, 0, nullptr, 0, 0,
      F(I_ENCB), xlin, 512, nullptr, 0, 512, 32, 1);
  xs_transpose_kernel<<<cdivu((unsigned)(NBS * 512), 256), 256, 0, stream>>>(xs0, xlin, S_, B_, 512);
  const unsigned BILSTM_SMEM = 32 * 1024 * 4 + 32 * 256 * 4 + 32 * 256 * 2;  // 176KB in 320KB LDS
  bilstm_layer_kernel<<<2, 256, BILSTM_SMEM, stream>>>(
      xs0, lWih[0], lWhh[0], lBg[0], lWih[1], lWhh[1], lBg[1], xs1, nullptr, S_, B_);
  bilstm_layer_kernel<<<2, 256, BILSTM_SMEM, stream>>>(
      xs1, lWih[2], lWhh[2], lBg[2], lWih[3], lWhh[3], lBg[3], memB, memF, S_, B_);
  // proc_mem = memory @ Wm^T + bm : [S*B, 128]
  gemm2_kernel<<<cdivu(512u * 8u, 4), 128, 0, stream>>>(
      memB, 512, mW, 512, 512, nullptr, 0, nullptr, 0, 0,
      F(I_MB), procM, 128, nullptr, 0, 512, 8, 0);

  // ---- decoder: 800 sequential steps ----
  for (int t = 0; t < T_; ++t) {
    // prenet
    gemm2_kernel<<<cdivu(2u * 16u, 4), 128, 0, stream>>>(
        melInB + (size_t)t * MELP_, T_ * MELP_, pre1W, MELP_, MELP_,
        nullptr, 0, nullptr, 0, 0, F(I_PRE1B), nullptr, 0, p1B, 256, 2, 16, 1);
    gemm2_kernel<<<cdivu(2u * 8u, 4), 128, 0, stream>>>(
        p1B, 256, pre2W, 256, 256, nullptr, 0, nullptr, 0, 0,
        F(I_PRE2B), nullptr, 0, attIn, 640, 2, 8, 1);
    // attention LSTM cell
    gemm2_kernel<<<cdivu(2u * 256u, 4), 128, 0, stream>>>(
        attIn, 640, attWih, 640, 640, decIn, 1536, attWhh, 1024, 1024,
        attBg, gates, 4096, nullptr, 0, 2, 256, 0);
    lstm_update_kernel<<<cdivu(32u * 1024u, 256), 256, 0, stream>>>(gates, cAtt, decIn, 1536, 1024, 32);
    // attention
    gemm2_kernel<<<cdivu(2u * 8u, 4), 128, 0, stream>>>(
        decIn, 1536, qW, 1024, 1024, nullptr, 0, nullptr, 0, 0,
        F(I_QB), qB, 128, nullptr, 0, 2, 8, 0);
    att_energy_kernel<<<B_ * S_, 128, 0, stream>>>(
        qB, procM, aprev, F(I_LOCW), F(I_LOCB), F(I_VW), F(I_VB), energ, S_, B_);
    softmax_ctx_kernel<<<B_, 256, 0, stream>>>(energ, memF, aprev, attIn, decIn, projIn, S_, B_);
    // decoder LSTM cell
    gemm2_kernel<<<cdivu(2u * 256u, 4), 128, 0, stream>>>(
        decIn, 1536, decWih, 1536, 1536, projIn, 1536, decWhh, 1024, 1024,
        decBg, gates, 4096, nullptr, 0, 2, 256, 0);
    lstm_update_kernel<<<cdivu(32u * 1024u, 256), 256, 0, stream>>>(gates, cDec, projIn, 1536, 1024, 32);
    // projections (mel frame -> d_out + postnet input; stop token)
    gemm2_kernel<<<cdivu(2u * 5u, 4), 128, 0, stream>>>(
        projIn, 1536, melW, 1536, 1536, nullptr, 0, nullptr, 0, 0,
        F(I_MELB), out + (size_t)t * MEL_, T_ * MEL_,
        Xp0 + (size_t)(t + 2) * MELP_, (T_ + 4) * MELP_, 2, 5, 0);
    stop_kernel<<<1, 32, 0, stream>>>(projIn, F(I_STOPW), F(I_STOPB),
                                      out + (size_t)2 * B_ * T_ * MEL_, t, T_);
  }

  // ---- postnet ----
  unsigned tiles512 = (unsigned)(B_ * (T_ / 16) * (512 / 16));
  unsigned tiles80  = (unsigned)(B_ * (T_ / 16) * (MEL_ / 16));
  postnet_conv_kernel<<<cdivu(tiles512, 4), 128, 0, stream>>>(
      Xp0, MELP_, postW[0], postSc[0], postSh[0], Xp1, 512, nullptr, nullptr, B_, T_, 512, 1);
  postnet_conv_kernel<<<cdivu(tiles512, 4), 128, 0, stream>>>(
      Xp1, 512, postW[1], postSc[1], postSh[1], Xp2, 512, nullptr, nullptr, B_, T_, 512, 1);
  postnet_conv_kernel<<<cdivu(tiles512, 4), 128, 0, stream>>>(
      Xp2, 512, postW[2], postSc[2], postSh[2], Xp1, 512, nullptr, nullptr, B_, T_, 512, 1);
  postnet_conv_kernel<<<cdivu(tiles512, 4), 128, 0, stream>>>(
      Xp1, 512, postW[3], postSc[3], postSh[3], Xp2, 512, nullptr, nullptr, B_, T_, 512, 1);
  postnet_conv_kernel<<<cdivu(tiles80, 4), 128, 0, stream>>>(
      Xp2, 512, postW[4], postSc[4], postSh[4], nullptr, 0,
      out + (size_t)B_ * T_ * MEL_, out, B_, T_, MEL_, 0);
}
